// SA_Layer_61211873903149
// MI455X (gfx1250) — matrix operand order repaired and checked by reading, not compiled, not run on hardware
//
#include <hip/hip_runtime.h>
#include <hip/hip_bf16.h>
#include <math.h>

// ---------------------------------------------------------------------------
// SA layer: q=Wq x, k=Wk x, v=Wv x+bv ; E=q^T k ; A=softmax_m(E) ; out=x+v A
//   B=4, C=256, N=4096, CQ=64.  (MI455X / gfx1250, wave32, bf16 WMMA, f32 acc)
//
//   K1: projection GEMMs -> qT [B,N,64] bf16, kT [B,N,64] bf16 (transposed so
//       B-fragments are contiguous b128 loads), v [B,256,N] bf16. ~12MB in
//       d_ws, L2-resident (192MB L2).
//   K2: per-row softmax stats: stream E=qT*kT^T through WMMA once, online
//       max/sum per row (energy tensor never touches HBM).
//   K3: out tile: recompute E tiles, p=exp(e-max)*rcp(sum) -> bf16 staged
//       transposed in LDS ([m][n], padded stride => b128 ds ops, bank-clean),
//       second WMMA v*p, epilogue += x.
// ---------------------------------------------------------------------------

typedef __attribute__((ext_vector_type(16))) __bf16 v16bf;
typedef __attribute__((ext_vector_type(8)))  __bf16 v8bf;
typedef __attribute__((ext_vector_type(8)))  float  v8f;
typedef __attribute__((ext_vector_type(4)))  float  v4f;

static constexpr int BB = 4;
static constexpr int CC = 256;
static constexpr int NN = 4096;
static constexpr int CQ = 64;

__device__ __forceinline__ v8f wmma_bf16(v16bf a, v16bf b, v8f c) {
  return __builtin_amdgcn_wmma_f32_16x16x32_bf16(
      false, a, false, b, (short)0, c, false, false);
}

__device__ __forceinline__ v16bf cat8(v8bf a, v8bf b) {
  return __builtin_shufflevector(a, b, 0, 1, 2, 3, 4, 5, 6, 7,
                                 8, 9, 10, 11, 12, 13, 14, 15);
}

// A-frag (16x32 bf16), row-major source row: per-lane halves live at
// K = hi*16 + [0..16)  -> two b128 loads.
__device__ __forceinline__ v16bf load_afrag(const __bf16* __restrict__ row,
                                            int hi) {
  v8bf lo = *(const v8bf*)(row + hi * 16);
  v8bf h8 = *(const v8bf*)(row + hi * 16 + 8);
  return cat8(lo, h8);
}

// B-frag (32x16 bf16), K-contiguous source (col-major matrix stored as rows of
// K): per-lane halves at K = hi*16 + [0..16)  -> two b128 loads.
__device__ __forceinline__ v16bf load_bfrag(const __bf16* __restrict__ col,
                                            int hi) {
  v8bf lo = *(const v8bf*)(col + hi * 16);
  v8bf h8 = *(const v8bf*)(col + hi * 16 + 8);
  return cat8(lo, h8);
}

// ---------------------------------------------------------------------------
// Kernel 1: projections. blockIdx.y (SGPR-uniform) picks the 16-row tile:
// [0,4)=q, [4,8)=k, [8,24)=v. Waves span 8 column tiles -> grid (N/128,24,B).
// ---------------------------------------------------------------------------
__global__ __launch_bounds__(256)
void proj_kernel(const float* __restrict__ x,  const float* __restrict__ Wq,
                 const float* __restrict__ Wk, const float* __restrict__ Wv,
                 const float* __restrict__ bv,
                 __bf16* __restrict__ qT, __bf16* __restrict__ kT,
                 __bf16* __restrict__ vp)
{
  const int b    = blockIdx.z;
  const int lane = threadIdx.x & 31;
  const int wave = threadIdx.x >> 5;
  const int hi   = lane >> 4;
  const int l15  = lane & 15;
  const int R0   = blockIdx.y * 16;                 // uniform (SGPR)
  const int n0   = (blockIdx.x * 8 + wave) * 16;

  const float* W;
  int lr0;
  if (R0 < 64)       { W = Wq; lr0 = R0; }
  else if (R0 < 128) { W = Wk; lr0 = R0 - 64; }
  else               { W = Wv; lr0 = R0 - 128; }
  const float* Wrow = W + (size_t)(lr0 + l15) * CC;

  const float* xb = x + (size_t)b * CC * NN;
  v8f acc = {};
  for (int kc = 0; kc < CC; kc += 32) {
    // A: weight tile (f32 -> bf16), two 8-float contiguous groups per lane
    v4f w0 = *(const v4f*)(Wrow + kc + hi * 8);
    v4f w1 = *(const v4f*)(Wrow + kc + hi * 8 + 4);
    v4f w2 = *(const v4f*)(Wrow + kc + hi * 8 + 16);
    v4f w3 = *(const v4f*)(Wrow + kc + hi * 8 + 20);
    v16bf a;
#pragma unroll
    for (int j = 0; j < 4; ++j) {
      a[j]      = (__bf16)w0[j];
      a[j + 4]  = (__bf16)w1[j];
      a[j + 8]  = (__bf16)w2[j];
      a[j + 12] = (__bf16)w3[j];
    }
    // B: x tile (column n0+l15, K strided by N) -> b32 gathers
    v16bf bf;
#pragma unroll
    for (int j = 0; j < 16; ++j)
      bf[j] = (__bf16)xb[(size_t)(kc + hi * 8 + j + (j >= 8 ? 8 : 0)) * NN + (n0 + l15)];
    acc = wmma_bf16(a, bf, acc);
  }

  const int n = n0 + l15;                           // D: lane=col, vgpr=row
  if (R0 < 128) {                                   // q / k: transposed [N,64]
    v8bf st;
#pragma unroll
    for (int r = 0; r < 8; ++r) st[r] = (__bf16)acc[r];
    __bf16* dst = (R0 < 64) ? qT : kT;
    *(v8bf*)(dst + ((size_t)b * NN + n) * CQ + lr0 + hi * 8) = st;
  } else {                                          // v: [256,N] + bias
#pragma unroll
    for (int r = 0; r < 8; ++r) {
      const int row = lr0 + hi * 8 + r;
      vp[((size_t)b * CC + row) * NN + n] = (__bf16)(acc[r] + bv[row]);
    }
  }
}

// ---------------------------------------------------------------------------
// Kernel 2: per-row softmax stats over all N=4096 m. grid (N/128, B),
// block 256; wave owns 16 energy rows, streams 256 m-tiles.
// ---------------------------------------------------------------------------
__global__ __launch_bounds__(256)
void stats_kernel(const __bf16* __restrict__ qT, const __bf16* __restrict__ kT,
                  float* __restrict__ rowmax, float* __restrict__ rowsum)
{
  const int b    = blockIdx.y;
  const int lane = threadIdx.x & 31;
  const int wave = threadIdx.x >> 5;
  const int hi   = lane >> 4;
  const int l15  = lane & 15;
  const int n0   = (blockIdx.x * 8 + wave) * 16;

  const __bf16* qrow = qT + ((size_t)b * NN + (n0 + l15)) * CQ;
  const v16bf aq0 = load_afrag(qrow, hi);
  const v16bf aq1 = load_afrag(qrow + 32, hi);

  float macc[8], sacc[8];
#pragma unroll
  for (int r = 0; r < 8; ++r) { macc[r] = -3.0e38f; sacc[r] = 0.f; }

  const __bf16* kb0 = kT + (size_t)b * NN * CQ;
  for (int mt = 0; mt < NN / 16; ++mt) {
    const __bf16* krow = kb0 + (size_t)(mt * 16 + l15) * CQ;
    __builtin_prefetch(krow + 16 * CQ, 0, 1);
    const v16bf kf0 = load_bfrag(krow, hi);
    const v16bf kf1 = load_bfrag(krow + 32, hi);
    v8f e = {};
    e = wmma_bf16(aq0, kf0, e);
    e = wmma_bf16(aq1, kf1, e);
#pragma unroll
    for (int r = 0; r < 8; ++r) {                   // online max/sum
      const float ev = e[r];
      const float mn = fmaxf(macc[r], ev);
      sacc[r] = sacc[r] * __expf(macc[r] - mn) + __expf(ev - mn);
      macc[r] = mn;
    }
  }
#pragma unroll
  for (int mask = 1; mask < 16; mask <<= 1) {       // 16-lane merge
#pragma unroll
    for (int r = 0; r < 8; ++r) {
      const float om = __shfl_xor(macc[r], mask, 32);
      const float os = __shfl_xor(sacc[r], mask, 32);
      const float mn = fmaxf(macc[r], om);
      sacc[r] = sacc[r] * __expf(macc[r] - mn) + os * __expf(om - mn);
      macc[r] = mn;
    }
  }
  if (l15 == 0) {
#pragma unroll
    for (int r = 0; r < 8; ++r) {
      const int n = n0 + hi * 8 + r;
      rowmax[(size_t)b * NN + n] = macc[r];
      rowsum[(size_t)b * NN + n] = sacc[r];
    }
  }
}

// ---------------------------------------------------------------------------
// Kernel 3: out[c,m] = x[c,m] + sum_n v[c,n]*exp(E[n,m]-max[n])/sum[n]
// grid (N/32, B), block 256. LDS: P^T [32 m][256 n] bf16, padded row stride
// 264 halves (132 dwords -> lanes 4 banks apart, b128 ds ops conflict-free).
// ---------------------------------------------------------------------------
static constexpr int PSTR = 264;                     // padded n-stride (halves)

__global__ __launch_bounds__(256)
void attn_out_kernel(const float* __restrict__ x,
                     const __bf16* __restrict__ qT, const __bf16* __restrict__ kT,
                     const __bf16* __restrict__ vp,
                     const float* __restrict__ rowmax,
                     const float* __restrict__ rowsum,
                     float* __restrict__ out)
{
  __shared__ __bf16 pbufT[32 * PSTR];                // [m_local][n_local]
  const int b    = blockIdx.y;
  const int lane = threadIdx.x & 31;
  const int wave = threadIdx.x >> 5;
  const int hi   = lane >> 4;
  const int l15  = lane & 15;
  const int m0   = blockIdx.x * 32;

  // k B-frags for both m-tiles: b128 loads, reused for all N/256 iterations
  v16bf kf[2][2];
  {
    const __bf16* kb0 = kT + (size_t)b * NN * CQ;
#pragma unroll
    for (int t = 0; t < 2; ++t) {
      const __bf16* krow = kb0 + (size_t)(m0 + t * 16 + l15) * CQ;
      kf[t][0] = load_bfrag(krow, hi);
      kf[t][1] = load_bfrag(krow + 32, hi);
    }
  }

  v8f acc[2][2] = {};                                // [ctile][mtile]
  const float*  rm = rowmax + (size_t)b * NN;
  const float*  rs = rowsum + (size_t)b * NN;
  const __bf16* vb = vp + (size_t)b * CC * NN;

  for (int nb = 0; nb < NN; nb += 256) {
    // ---- phase 1: P for this wave's 32-row n-subchunk, both m-tiles
    const int nw = nb + wave * 32;
#pragma unroll
    for (int nt = 0; nt < 2; ++nt) {
      const __bf16* qrow = qT + ((size_t)b * NN + (nw + nt * 16 + l15)) * CQ;
      const v16bf aq0 = load_afrag(qrow, hi);
      const v16bf aq1 = load_afrag(qrow + 32, hi);
      float rmv[8], rsi[8];
#pragma unroll
      for (int r = 0; r < 8; ++r) {                  // stats for this n-tile
        const int n = nw + nt * 16 + hi * 8 + r;
        rmv[r] = rm[n];
        rsi[r] = __builtin_amdgcn_rcpf(rs[n]);
      }
#pragma unroll
      for (int mt = 0; mt < 2; ++mt) {
        v8f e = {};
        e = wmma_bf16(aq0, kf[mt][0], e);
        e = wmma_bf16(aq1, kf[mt][1], e);
        v8bf pv;                                     // 8 contiguous n values
#pragma unroll
        for (int r = 0; r < 8; ++r)
          pv[r] = (__bf16)(__expf(e[r] - rmv[r]) * rsi[r]);
        *(v8bf*)&pbufT[(mt * 16 + l15) * PSTR +
                       wave * 32 + nt * 16 + hi * 8] = pv;  // ds_store_b128
      }
    }
    __syncthreads();

    // ---- phase 2: consume all 256 staged n rows
#pragma unroll 1
    for (int s = 0; s < 8; ++s) {
      v16bf pf[2];
#pragma unroll
      for (int mt = 0; mt < 2; ++mt) {               // two ds_load_b128 each
        const __bf16* prow = &pbufT[(mt * 16 + l15) * PSTR + s * 32];
        pf[mt] = cat8(*(const v8bf*)(prow + hi * 16),
                      *(const v8bf*)(prow + hi * 16 + 8));
      }
#pragma unroll
      for (int ct = 0; ct < 2; ++ct) {
        const __bf16* vrow =
            vb + (size_t)(wave * 32 + ct * 16 + l15) * NN + nb + s * 32;
        const v16bf av = load_afrag(vrow, hi);       // two global b128
#pragma unroll
        for (int mt = 0; mt < 2; ++mt)
          acc[ct][mt] = wmma_bf16(av, pf[mt], acc[ct][mt]);
      }
    }
    __syncthreads();
  }

  // ---- epilogue: out = x + acc
  const float* xb = x   + (size_t)b * CC * NN;
  float*       ob = out + (size_t)b * CC * NN;
#pragma unroll
  for (int ct = 0; ct < 2; ++ct)
#pragma unroll
    for (int mt = 0; mt < 2; ++mt) {
      const int m = m0 + mt * 16 + l15;
#pragma unroll
      for (int r = 0; r < 8; ++r) {
        const int c = wave * 32 + ct * 16 + hi * 8 + r;
        ob[(size_t)c * NN + m] = xb[(size_t)c * NN + m] + acc[ct][mt][r];
      }
    }
}

// ---------------------------------------------------------------------------
extern "C" void kernel_launch(void* const* d_in, const int* in_sizes, int n_in,
                              void* d_out, int out_size, void* d_ws,
                              size_t ws_size, hipStream_t stream)
{
  const float* x  = (const float*)d_in[0];
  const float* Wq = (const float*)d_in[1];
  const float* Wk = (const float*)d_in[2];
  const float* Wv = (const float*)d_in[3];
  const float* bv = (const float*)d_in[4];
  float* out = (float*)d_out;

  // workspace layout (~12.6 MB)
  __bf16* qT = (__bf16*)d_ws;                            // B*N*CQ
  __bf16* kT = qT + (size_t)BB * NN * CQ;                // B*N*CQ (transposed)
  __bf16* vp = kT + (size_t)BB * NN * CQ;                // B*C*N
  float* rowmax = (float*)(vp + (size_t)BB * CC * NN);   // B*N
  float* rowsum = rowmax + (size_t)BB * NN;              // B*N

  proj_kernel<<<dim3(NN / 128, 24, BB), 256, 0, stream>>>(
      x, Wq, Wk, Wv, bv, qT, kT, vp);
  stats_kernel<<<dim3(NN / 128, BB), 256, 0, stream>>>(
      qT, kT, rowmax, rowsum);
  attn_out_kernel<<<dim3(NN / 32, BB), 256, 0, stream>>>(
      x, qT, kT, vp, rowmax, rowsum, out);
}
